// pLinear_11012296146992
// MI455X (gfx1250) — compile-verified, matrix operand order checked by reading, not run on hardware
//
#include <hip/hip_runtime.h>
#include <hip/hip_bf16.h>
#include <math.h>

// ---------------------------------------------------------------------------
// y[n,h] = exp(log_scale[h]) * sum_d x[n,d]*Wint[h,d] + bias[h] * sum_d x[n,d]
//
// GEMM in bf16 WMMA with f32 accumulate. Wint (0..15) is EXACT in bf16; x is
// split x = x_hi + x_lo (two bf16 limbs) so the product is f32-accurate while
// still running on the matrix pipe:  y_gemm = x_hi@W^T + x_lo@W^T.
// Tiles stream global->LDS with async-to-LDS (ASYNCcnt) + double buffering.
// ---------------------------------------------------------------------------

#define N_TOK 2048
#define D_IN  4096
#define H_OUT 4096

#define TILE_M 128
#define TILE_N 128
#define KSTEP  32

typedef __bf16 bf16_t;
typedef __attribute__((ext_vector_type(16))) __bf16 v16bf;
typedef __attribute__((ext_vector_type(8)))  __bf16 v8bf;
typedef __attribute__((ext_vector_type(8)))  float  v8f;
typedef __attribute__((ext_vector_type(4)))  int    v4i;

#if __has_builtin(__builtin_amdgcn_global_load_async_to_lds_b128)
#define HAVE_ASYNC_LDS 1
typedef __attribute__((address_space(1))) v4i v4i_glb;
typedef __attribute__((address_space(3))) v4i v4i_lds;
#else
#define HAVE_ASYNC_LDS 0
#endif

// copy 32 bytes (16 bf16) global -> LDS for this lane
__device__ __forceinline__ void copy32B_to_lds(const bf16_t* g, bf16_t* l) {
#if HAVE_ASYNC_LDS
    v4i_glb* gp = (v4i_glb*)(unsigned long long)g;
    v4i_lds* lp = (v4i_lds*)(unsigned long long)l;
    __builtin_amdgcn_global_load_async_to_lds_b128(gp, lp, 0, 0);
    __builtin_amdgcn_global_load_async_to_lds_b128(gp, lp, 16, 0);
#else
    const uint4* gs = (const uint4*)g;
    uint4* ls = (uint4*)l;
    ls[0] = gs[0];
    ls[1] = gs[1];
#endif
}

template <int N>
__device__ __forceinline__ void wait_async() {
#if HAVE_ASYNC_LDS
#if __has_builtin(__builtin_amdgcn_s_wait_asynccnt)
    __builtin_amdgcn_s_wait_asynccnt(N);
#else
    asm volatile("s_wait_asynccnt %0" ::"i"(N) : "memory");
#endif
#endif
}

// ---------------- x -> (hi, lo) bf16 limb split ----------------------------
__global__ void split_x_kernel(const float* __restrict__ src,
                               bf16_t* __restrict__ hi,
                               bf16_t* __restrict__ lo, int n) {
    int i = (blockIdx.x * blockDim.x + threadIdx.x) * 4;
    if (i + 3 < n) {
        float4 v = *(const float4*)(src + i);
        float vv[4] = {v.x, v.y, v.z, v.w};
#pragma unroll
        for (int j = 0; j < 4; ++j) {
            bf16_t h = (bf16_t)vv[j];
            hi[i + j] = h;
            lo[i + j] = (bf16_t)(vv[j] - (float)h);
        }
    }
}

// ---------------- Wint f32 -> bf16 (exact, values 0..15) -------------------
__global__ void cvt_w_kernel(const float* __restrict__ src,
                             bf16_t* __restrict__ dst, int n) {
    int i = (blockIdx.x * blockDim.x + threadIdx.x) * 4;
    if (i + 3 < n) {
        float4 v = *(const float4*)(src + i);
        dst[i + 0] = (bf16_t)v.x;
        dst[i + 1] = (bf16_t)v.y;
        dst[i + 2] = (bf16_t)v.z;
        dst[i + 3] = (bf16_t)v.w;
    }
}

// ---------------- per-token row sum of x -----------------------------------
__global__ void rowsum_kernel(const float* __restrict__ x,
                              float* __restrict__ rs) {
    __shared__ float red[256];
    const int row = blockIdx.x;
    float s = 0.f;
    for (int c = threadIdx.x; c < D_IN; c += 256)
        s += x[(size_t)row * D_IN + c];
    red[threadIdx.x] = s;
    __syncthreads();
    for (int w = 128; w > 0; w >>= 1) {
        if ((int)threadIdx.x < w) red[threadIdx.x] += red[threadIdx.x + w];
        __syncthreads();
    }
    if (threadIdx.x == 0) rs[row] = red[0];
}

// ---------------- WMMA GEMM + fused dequant epilogue -----------------------
// Block: 256 threads = 8 wave32s; tile 128(M) x 128(N); K-step 32.
// Waves 4(M) x 2(N); each wave: 2x4 WMMA 16x16 tiles, 16 wmma per K-step
// (hi and lo A-limbs share the B fragments).
__global__ __launch_bounds__(256)
void gemm_bf16_wmma_kernel(const bf16_t* __restrict__ Ahi,  // [N_TOK, D_IN]
                           const bf16_t* __restrict__ Alo,  // [N_TOK, D_IN]
                           const bf16_t* __restrict__ B,    // [H_OUT, D_IN]
                           const float* __restrict__ log_scale,
                           const float* __restrict__ bias,
                           const float* __restrict__ rowsum,
                           float* __restrict__ out)         // [N_TOK, H_OUT]
{
    __shared__ __align__(16) bf16_t sAh[2][TILE_M * KSTEP];  // 2 x 8 KB
    __shared__ __align__(16) bf16_t sAl[2][TILE_M * KSTEP];  // 2 x 8 KB
    __shared__ __align__(16) bf16_t sB [2][TILE_N * KSTEP];  // 2 x 8 KB

    const int m_base = blockIdx.y * TILE_M;
    const int h_base = blockIdx.x * TILE_N;
    const int tid    = threadIdx.x;
    const int lane   = tid & 31;
    const int wave   = tid >> 5;
    const int waveM  = wave & 3;              // M offset waveM*32
    const int waveN  = wave >> 2;             // N offset waveN*64
    const int lrow   = lane & 15;
    const int khalf  = (lane >> 4) ? 8 : 0;   // ISA bf16 fragment K-half

    // cooperative copy slice: 2 threads per 32-col row, 32 B each
    const int crow = tid >> 1;
    const int coff = (tid & 1) * 16;

    v8f acc[2][4];
#pragma unroll
    for (int tm = 0; tm < 2; ++tm)
#pragma unroll
        for (int tn = 0; tn < 4; ++tn)
#pragma unroll
            for (int r = 0; r < 8; ++r) acc[tm][tn][r] = 0.f;

    auto issue_tiles = [&](int buf, int k0) {
        copy32B_to_lds(Ahi + (size_t)(m_base + crow) * D_IN + k0 + coff,
                       &sAh[buf][crow * KSTEP + coff]);
        copy32B_to_lds(Alo + (size_t)(m_base + crow) * D_IN + k0 + coff,
                       &sAl[buf][crow * KSTEP + coff]);
        copy32B_to_lds(B + (size_t)(h_base + crow) * D_IN + k0 + coff,
                       &sB[buf][crow * KSTEP + coff]);
    };

    const int nIter = D_IN / KSTEP;
    issue_tiles(0, 0);   // prologue prefetch

    for (int it = 0; it < nIter; ++it) {
        const int cur = it & 1;
        const int k0  = it * KSTEP;

        // prefetch next tile into the other buffer (safe: other buffer's
        // readers all passed the previous end-of-iteration barrier)
        if (it + 1 < nIter) {
            issue_tiles(cur ^ 1, k0 + KSTEP);
            wait_async<6>();   // in-order: current buffer's 6 loads landed
        } else {
            wait_async<0>();
        }
        __syncthreads();       // everyone's current-buffer data visible

        v16bf ahi[2], alo[2], bfrag[4];
#pragma unroll
        for (int tm = 0; tm < 2; ++tm) {
            const int mr = waveM * 32 + tm * 16 + lrow;
            v8bf lo = *(const v8bf*)(&sAh[cur][mr * KSTEP + khalf]);
            v8bf hi = *(const v8bf*)(&sAh[cur][mr * KSTEP + khalf + 16]);
            ahi[tm] = __builtin_shufflevector(lo, hi, 0, 1, 2, 3, 4, 5, 6, 7,
                                              8, 9, 10, 11, 12, 13, 14, 15);
            v8bf lo2 = *(const v8bf*)(&sAl[cur][mr * KSTEP + khalf]);
            v8bf hi2 = *(const v8bf*)(&sAl[cur][mr * KSTEP + khalf + 16]);
            alo[tm] = __builtin_shufflevector(lo2, hi2, 0, 1, 2, 3, 4, 5, 6, 7,
                                              8, 9, 10, 11, 12, 13, 14, 15);
        }
#pragma unroll
        for (int tn = 0; tn < 4; ++tn) {
            const int nr = waveN * 64 + tn * 16 + lrow;
            v8bf lo = *(const v8bf*)(&sB[cur][nr * KSTEP + khalf]);
            v8bf hi = *(const v8bf*)(&sB[cur][nr * KSTEP + khalf + 16]);
            bfrag[tn] = __builtin_shufflevector(lo, hi, 0, 1, 2, 3, 4, 5, 6, 7,
                                                8, 9, 10, 11, 12, 13, 14, 15);
        }

#pragma unroll
        for (int tm = 0; tm < 2; ++tm)
#pragma unroll
            for (int tn = 0; tn < 4; ++tn) {
                acc[tm][tn] = __builtin_amdgcn_wmma_f32_16x16x32_bf16(
                    false, ahi[tm], false, bfrag[tn],
                    (short)0, acc[tm][tn], false, false);
                acc[tm][tn] = __builtin_amdgcn_wmma_f32_16x16x32_bf16(
                    false, alo[tm], false, bfrag[tn],
                    (short)0, acc[tm][tn], false, false);
            }

        __syncthreads();   // all reads of current buffer done before reuse
    }

    // Epilogue: y = exp(log_scale[h]) * acc + bias[h] * rowsum[m]
    const int rowadd = (lane >> 4) * 8;   // C/D layout: lanes 16-31 are M+8
    float rs[2][8];
#pragma unroll
    for (int tm = 0; tm < 2; ++tm)
#pragma unroll
        for (int r = 0; r < 8; ++r)
            rs[tm][r] = rowsum[m_base + waveM * 32 + tm * 16 + rowadd + r];

#pragma unroll
    for (int tn = 0; tn < 4; ++tn) {
        const int h = h_base + waveN * 64 + tn * 16 + lrow;
        const float sc = __expf(log_scale[h]);
        const float bi = bias[h];
#pragma unroll
        for (int tm = 0; tm < 2; ++tm) {
            const int m0 = m_base + waveM * 32 + tm * 16 + rowadd;
#pragma unroll
            for (int r = 0; r < 8; ++r) {
                out[(size_t)(m0 + r) * H_OUT + h] =
                    sc * acc[tm][tn][r] + bi * rs[tm][r];
            }
        }
    }
}

// ---------------------------------------------------------------------------
extern "C" void kernel_launch(void* const* d_in, const int* in_sizes, int n_in,
                              void* d_out, int out_size, void* d_ws, size_t ws_size,
                              hipStream_t stream) {
    const float* x         = (const float*)d_in[0];   // [N_TOK, D_IN]
    const float* log_scale = (const float*)d_in[1];   // [H_OUT]
    const float* bias      = (const float*)d_in[2];   // [H_OUT]
    const float* Wint      = (const float*)d_in[3];   // [H_OUT, D_IN], 0..15
    // d_in[4]=W, d_in[5]=L, d_in[6]=nbits : unused by the reference math
    float* out = (float*)d_out;

    // workspace: xhi (16MB) | xlo (16MB) | wb (32MB) | rowsum (8KB)
    bf16_t* xhi = (bf16_t*)d_ws;
    bf16_t* xlo = xhi + (size_t)N_TOK * D_IN;
    bf16_t* wb  = xlo + (size_t)N_TOK * D_IN;
    float*  rs  = (float*)(wb + (size_t)H_OUT * D_IN);

    {
        int n = N_TOK * D_IN;
        split_x_kernel<<<(n / 4 + 255) / 256, 256, 0, stream>>>(x, xhi, xlo, n);
    }
    {
        int n = H_OUT * D_IN;
        cvt_w_kernel<<<(n / 4 + 255) / 256, 256, 0, stream>>>(Wint, wb, n);
    }
    rowsum_kernel<<<N_TOK, 256, 0, stream>>>(x, rs);

    dim3 grid(H_OUT / TILE_N, N_TOK / TILE_M);   // 32 x 16 = 512 blocks
    gemm_bf16_wmma_kernel<<<grid, 256, 0, stream>>>(xhi, xlo, wb, log_scale,
                                                    bias, rs, out);
}
